// AttentionHead_11106785427959
// MI455X (gfx1250) — compile-verified, matrix operand order checked by reading, not run on hardware
//
#include <hip/hip_runtime.h>
#include <hip/hip_bf16.h>

// ---------------------------------------------------------------------------
// AttentionHead for MI455X (gfx1250, wave32, WMMA f16 16x16x32 -> f32).
// Phase 1: QKV projection  x[BT,E] @ W[E,H] + b  -> f16 in workspace.
// Phase 2: fused flash attention (online softmax) -> f32 out.
// ---------------------------------------------------------------------------

#define B_  8
#define T_  2048
#define E_  1024
#define H_  64
#define BT_ (B_ * T_)

typedef _Float16 h2  __attribute__((ext_vector_type(2)));
typedef _Float16 h4  __attribute__((ext_vector_type(4)));
typedef _Float16 h8  __attribute__((ext_vector_type(8)));
typedef _Float16 h16 __attribute__((ext_vector_type(16)));
typedef float    f4  __attribute__((ext_vector_type(4)));
typedef float    f8  __attribute__((ext_vector_type(8)));

static __device__ __forceinline__ h16 hcat(h8 a, h8 b) {
    return __builtin_shufflevector(a, b, 0,1,2,3,4,5,6,7,8,9,10,11,12,13,14,15);
}

// D = A(16x32 f16) x B(32x16 f16) + C(16x16 f32)
static __device__ __forceinline__ f8 wmma_f16(h16 a, h16 b, f8 c) {
    return __builtin_amdgcn_wmma_f32_16x16x32_f16(
        /*neg_a=*/false, a, /*neg_b=*/false, b,
        /*c_mod=*/(short)0, c, /*reuse_a=*/false, /*reuse_b=*/false);
}

static __device__ __forceinline__ void lds_fence() {
    __builtin_amdgcn_wave_barrier();
    asm volatile("s_wait_dscnt 0x0" ::: "memory");
    __builtin_amdgcn_wave_barrier();
}

// ---------------------------------------------------------------------------
// Kernel 1: QKV projection.  grid = (BT/128, 1, 3), block = 256 (8 waves).
// Each block: 128x64 output tile of Q (z=0), K (z=1) or V (z=2).
// K staged 64 wide (2 WMMA k-steps per barrier pair, 8 WMMAs per stage).
// A fragment (per lane, 16 halves): lanes 0-15 row M=l, K={0..7,16..23};
// lanes 16-31 row M=l-16, K={8..15,24..31}.  B mirrors with lane = column.
// ---------------------------------------------------------------------------
__global__ __launch_bounds__(256)
void qkv_proj_kernel(const float* __restrict__ x,
                     const float* __restrict__ Wq, const float* __restrict__ bq,
                     const float* __restrict__ Wk, const float* __restrict__ bk,
                     const float* __restrict__ Wv, const float* __restrict__ bv,
                     _Float16* __restrict__ qkv)
{
    constexpr int KC   = 64;   // K per stage
    constexpr int STRA = 72;   // halves per A row in LDS (144 B, 16B aligned)
    constexpr int STRB = 72;   // halves per W^T row in LDS
    __shared__ _Float16 Alds[128 * STRA];
    __shared__ _Float16 Wlds[64 * STRB];

    const int z = blockIdx.z;
    const float* W    = (z == 0) ? Wq : (z == 1) ? Wk : Wv;
    const float* bias = (z == 0) ? bq : (z == 1) ? bk : bv;
    _Float16* out = qkv + (size_t)z * BT_ * H_;

    const int m0   = blockIdx.x * 128;
    const int t    = threadIdx.x;
    const int wave = t >> 5;
    const int lane = t & 31;
    const int ln   = lane & 15;
    const int hi8  = (lane >> 4) * 8;
    const int mrow = wave * 16 + ln;          // A-fragment row within tile

    f8 acc[4] = {};

    for (int kb = 0; kb < E_; kb += KC) {
        // prefetch next x stage (global_prefetch_b8)
        if (kb + KC < E_) {
            int u = t;
            int r = u >> 4, cv = (u & 15) * 4;
            __builtin_prefetch(x + (size_t)(m0 + r) * E_ + kb + KC + cv, 0, 1);
        }
        __syncthreads();
        // stage x tile (128 rows x 64 cols) as f16: 8 x f4 per thread
        #pragma unroll
        for (int i = 0; i < 8; ++i) {
            int u = t + i * 256;
            int r = u >> 4, cv = (u & 15) * 4;
            f4 xv = *(const f4*)(x + (size_t)(m0 + r) * E_ + kb + cv);
            h4 hv = { (_Float16)xv[0], (_Float16)xv[1],
                      (_Float16)xv[2], (_Float16)xv[3] };
            *(h4*)(Alds + r * STRA + cv) = hv;
        }
        // stage W tile (64 k x 64 cols) transposed as f16: Wlds[col][k]
        #pragma unroll
        for (int i = 0; i < 4; ++i) {
            int u = t + i * 256;
            int k = u >> 4, cv = (u & 15) * 4;
            f4 wv = *(const f4*)(W + (size_t)(kb + k) * H_ + cv);
            #pragma unroll
            for (int j = 0; j < 4; ++j)
                Wlds[(cv + j) * STRB + k] = (_Float16)wv[j];
        }
        __syncthreads();

        #pragma unroll
        for (int c = 0; c < 2; ++c) {
            const int ko = c * 32;
            // hoist all fragment loads, then issue WMMAs back-to-back
            h8 alo = *(const h8*)(Alds + mrow * STRA + ko + hi8);
            h8 ahi = *(const h8*)(Alds + mrow * STRA + ko + 16 + hi8);
            h16 a = hcat(alo, ahi);
            h16 bf[4];
            #pragma unroll
            for (int j = 0; j < 4; ++j) {
                int col = j * 16 + ln;
                h8 blo = *(const h8*)(Wlds + col * STRB + ko + hi8);
                h8 bhi = *(const h8*)(Wlds + col * STRB + ko + 16 + hi8);
                bf[j] = hcat(blo, bhi);
            }
            #pragma unroll
            for (int j = 0; j < 4; ++j)
                acc[j] = wmma_f16(a, bf[j], acc[j]);
        }
    }

    // bias add + f16 store.  D layout: VGPR r holds row r (lanes 0-15) or
    // row 8+r (lanes 16-31), col = j*16 + (lane&15).
    #pragma unroll
    for (int j = 0; j < 4; ++j) {
        int col = j * 16 + ln;
        float bj = bias[col];
        #pragma unroll
        for (int r = 0; r < 8; ++r) {
            int row = m0 + wave * 16 + r + hi8;
            out[(size_t)row * H_ + col] = (_Float16)(acc[j][r] + bj);
        }
    }
}

// ---------------------------------------------------------------------------
// Kernel 2: fused flash attention.  grid = (T/64, B), block = 128 (4 waves).
// Each wave owns 16 query rows; loop over key blocks of 32 with online softmax.
// ---------------------------------------------------------------------------
__global__ __launch_bounds__(128)
void flash_attn_kernel(const _Float16* __restrict__ Qh,
                       const _Float16* __restrict__ Kh,
                       const _Float16* __restrict__ Vh,
                       const int* __restrict__ mask,
                       float* __restrict__ out)
{
    constexpr int SSTR = 36;   // floats per S row in LDS
    constexpr int PSTR = 40;   // halves per P / V^T row in LDS
    __shared__ _Float16 VT[64 * PSTR];      // V^T: [col][key]
    __shared__ float    maskadd[32];
    __shared__ float    Sbuf[4][16 * SSTR]; // per-wave S scratch
    __shared__ _Float16 Pbuf[4][16 * PSTR]; // per-wave P (A-frag layout)
    __shared__ float    sc[4][16];          // per-wave row broadcast

    const int b    = blockIdx.y;
    const int q0   = blockIdx.x * 64;
    const int t    = threadIdx.x;
    const int wave = t >> 5;
    const int lane = t & 31;
    const int ln   = lane & 15;
    const int hi8  = (lane >> 4) * 8;

    const size_t kvbase = (size_t)b * T_ * H_;
    const int    qrow   = q0 + wave * 16 + ln;
    const size_t qbase  = kvbase + (size_t)qrow * H_;

    // preload Q fragments, pre-scaled by H^-0.5 = 0.125 (exact in f16)
    h16 aq[2];
    #pragma unroll
    for (int c = 0; c < 2; ++c) {
        h8 lo = *(const h8*)(Qh + qbase + c * 32 + hi8);
        h8 hc = *(const h8*)(Qh + qbase + c * 32 + 16 + hi8);
        h16 a = hcat(lo, hc);
        #pragma unroll
        for (int i = 0; i < 16; ++i) a[i] = a[i] * (_Float16)0.125f;
        aq[c] = a;
    }

    f8 o[4] = {};
    float mrun = -1e30f, lrun = 0.0f;

    // V staging decomposition: thread t handles cols cg*8..cg*8+7 of keys
    // 2*kp, 2*kp+1 and writes packed key-pairs (h2) into VT[col][key].
    const int kp = t & 15;
    const int cg = t >> 4;

    for (int kb = 0; kb < T_ / 32; ++kb) {
        const int n0 = kb * 32;
        __syncthreads();                     // protect VT/maskadd rewrite
        {
            const _Float16* v0p = Vh + kvbase + (size_t)(n0 + 2 * kp) * H_ + cg * 8;
            h8 v0 = *(const h8*)(v0p);
            h8 v1 = *(const h8*)(v0p + H_);
            #pragma unroll
            for (int i = 0; i < 8; ++i) {
                h2 pr = { v0[i], v1[i] };
                *(h2*)(VT + (cg * 8 + i) * PSTR + 2 * kp) = pr;
            }
        }
        if (t < 32) maskadd[t] = mask[b * T_ + n0 + t] ? 0.0f : -1e9f;
        __syncthreads();

        // S = (Q*scale) @ K^T : hoist all 4 K fragments, then 4 WMMAs
        h16 kf[2][2];
        #pragma unroll
        for (int c = 0; c < 2; ++c)
            #pragma unroll
            for (int j = 0; j < 2; ++j) {
                size_t kbn = kvbase + (size_t)(n0 + j * 16 + ln) * H_;
                h8 lo = *(const h8*)(Kh + kbn + c * 32 + hi8);
                h8 hc = *(const h8*)(Kh + kbn + c * 32 + 16 + hi8);
                kf[c][j] = hcat(lo, hc);
            }
        f8 s0 = {}, s1 = {};
        s0 = wmma_f16(aq[0], kf[0][0], s0);
        s1 = wmma_f16(aq[0], kf[0][1], s1);
        s0 = wmma_f16(aq[1], kf[1][0], s0);
        s1 = wmma_f16(aq[1], kf[1][1], s1);

        // spill S to per-wave LDS scratch
        float* sw = Sbuf[wave];
        #pragma unroll
        for (int r = 0; r < 8; ++r) {
            int row = r + hi8;
            sw[row * SSTR + ln]      = s0[r];
            sw[row * SSTR + 16 + ln] = s1[r];
        }
        lds_fence();

        // online softmax: lane r (< 16) owns query row r of this wave
        if (lane < 16) {
            const float* srow = sw + lane * SSTR;
            float sv[32];
            float mx = -1e30f;
            #pragma unroll
            for (int k = 0; k < 32; ++k) {
                sv[k] = srow[k] + maskadd[k];
                mx = fmaxf(mx, sv[k]);
            }
            float mnew  = fmaxf(mrun, mx);
            float alpha = __expf(mrun - mnew);
            float sum = 0.0f;
            _Float16* prow = Pbuf[wave] + lane * PSTR;
            #pragma unroll
            for (int k = 0; k < 32; ++k) {
                float p = __expf(sv[k] - mnew);
                sum += p;
                prow[k] = (_Float16)p;
            }
            lrun = alpha * lrun + sum;
            mrun = mnew;
            sc[wave][lane] = alpha;
        }
        lds_fence();

        // rescale accumulator by alpha(row)
        f4 a0 = *(const f4*)&sc[wave][hi8];
        f4 a1 = *(const f4*)&sc[wave][hi8 + 4];
        float ar[8] = { a0[0], a0[1], a0[2], a0[3], a1[0], a1[1], a1[2], a1[3] };
        #pragma unroll
        for (int j = 0; j < 4; ++j)
            #pragma unroll
            for (int r = 0; r < 8; ++r)
                o[j][r] *= ar[r];

        // O += P @ V : hoist P fragment + 4 V fragments, then 4 WMMAs
        h8 plo = *(const h8*)(Pbuf[wave] + ln * PSTR + hi8);
        h8 phi = *(const h8*)(Pbuf[wave] + ln * PSTR + 16 + hi8);
        h16 pf = hcat(plo, phi);
        h16 vf[4];
        #pragma unroll
        for (int j = 0; j < 4; ++j) {
            int col = j * 16 + ln;
            h8 vlo = *(const h8*)(VT + col * PSTR + hi8);
            h8 vhi = *(const h8*)(VT + col * PSTR + 16 + hi8);
            vf[j] = hcat(vlo, vhi);
        }
        #pragma unroll
        for (int j = 0; j < 4; ++j)
            o[j] = wmma_f16(pf, vf[j], o[j]);
    }

    // final 1/l normalization + f32 store
    if (lane < 16) sc[wave][lane] = 1.0f / lrun;
    lds_fence();
    f4 i0 = *(const f4*)&sc[wave][hi8];
    f4 i1 = *(const f4*)&sc[wave][hi8 + 4];
    float ir[8] = { i0[0], i0[1], i0[2], i0[3], i1[0], i1[1], i1[2], i1[3] };

    #pragma unroll
    for (int j = 0; j < 4; ++j) {
        int col = j * 16 + ln;
        #pragma unroll
        for (int r = 0; r < 8; ++r) {
            int row = q0 + wave * 16 + r + hi8;
            out[((size_t)b * T_ + row) * H_ + col] = o[j][r] * ir[r];
        }
    }
}

// ---------------------------------------------------------------------------
// Launch
// ---------------------------------------------------------------------------
extern "C" void kernel_launch(void* const* d_in, const int* in_sizes, int n_in,
                              void* d_out, int out_size, void* d_ws, size_t ws_size,
                              hipStream_t stream)
{
    const float* x    = (const float*)d_in[0];
    const float* Wq   = (const float*)d_in[1];
    const float* bq   = (const float*)d_in[2];
    const float* Wk   = (const float*)d_in[3];
    const float* bk   = (const float*)d_in[4];
    const float* Wv   = (const float*)d_in[5];
    const float* bv   = (const float*)d_in[6];
    const int*   mask = (const int*)d_in[7];
    float* out = (float*)d_out;

    _Float16* qkv = (_Float16*)d_ws;                 // 3 * BT * H halves = 6 MB
    const _Float16* Qh = qkv;
    const _Float16* Kh = qkv + (size_t)BT_ * H_;
    const _Float16* Vh = qkv + (size_t)2 * BT_ * H_;

    dim3 g1(BT_ / 128, 1, 3);
    qkv_proj_kernel<<<g1, 256, 0, stream>>>(x, Wq, bq, Wk, bk, Wv, bv, qkv);

    dim3 g2(T_ / 64, B_, 1);
    flash_attn_kernel<<<g2, 128, 0, stream>>>(Qh, Kh, Vh, mask, out);
}
